// GINandPool_47699906789506
// MI455X (gfx1250) — matrix operand order repaired and checked by reading, not compiled, not run on hardware
//
#include <hip/hip_runtime.h>
#include <hip/hip_bf16.h>

#define NN 100000
#define EE 1000000
#define HH 64
#define GG 256
#define LL 12
#define OUTD 16

typedef __attribute__((ext_vector_type(16))) __bf16 v16bf;
typedef __attribute__((ext_vector_type(8)))  float  v8f;

// ---------------------------------------------------------------------------
// fill with zeros (float4 granularity)
__global__ void gin_fill0(float4* __restrict__ p, int n4) {
    int t = blockIdx.x * blockDim.x + threadIdx.x;
    if (t < n4) p[t] = make_float4(0.f, 0.f, 0.f, 0.f);
}

// ---------------------------------------------------------------------------
// degree: 1 atomic per edge
__global__ void gin_deg(const int* __restrict__ src, float* __restrict__ deg) {
    int t = blockIdx.x * blockDim.x + threadIdx.x;
    if (t < EE) unsafeAtomicAdd(&deg[src[t]], 1.0f);
}

// ---------------------------------------------------------------------------
// encoder: x = relu(deg * W_enc + b_enc), elementwise over N*H
__global__ void gin_enc(const float* __restrict__ deg, const float* __restrict__ Wenc,
                        const float* __restrict__ benc, float* __restrict__ x) {
    int t = blockIdx.x * blockDim.x + threadIdx.x;  // N*H exact multiple of 256
    int node = t >> 6, c = t & 63;
    float v = deg[node] * Wenc[c] + benc[c];
    x[t] = v > 0.f ? v : 0.f;
}

// ---------------------------------------------------------------------------
// shuffle W1/W2 (fp32, row-major KxN) into WMMA bf16 B-fragment order.
// frag element index: ((l*2+g01)*4096) + ((nt*2+ks)*32 + lane)*16 + i
// B layout (16-bit, 32x16): lane half=lane/16, n=lane%16; element i -> K=32*ks+8*half+i+(i>=8?8:0), N=16*nt+n
__global__ void gin_wprep(const float* __restrict__ W1, const float* __restrict__ W2,
                          __bf16* __restrict__ wfrag) {
    int t = blockIdx.x * blockDim.x + threadIdx.x;  // LL*2*4096 = 98304 exact
    int i = t & 15;
    int q = t >> 4;
    int lane = q & 31; q >>= 5;
    int ks = q & 1;    q >>= 1;
    int nt = q & 3;    q >>= 2;
    int g01 = q & 1;
    int l = q >> 1;
    int K = 32 * ks + 8 * (lane >> 4) + i + (i >= 8 ? 8 : 0);
    int Nc = 16 * nt + (lane & 15);
    const float* W = (g01 ? W2 : W1) + l * HH * HH;
    wfrag[t] = (__bf16)W[K * HH + Nc];
}

// ---------------------------------------------------------------------------
// scatter: agg[dst] += x[src]; thread = (edge, 4-col group); float4 gather + 4 HW f32 atomics
__global__ void gin_scatter(const int* __restrict__ ei, const float* __restrict__ x,
                            float* __restrict__ agg) {
    int t = blockIdx.x * blockDim.x + threadIdx.x;  // EE*16 exact multiple of 256
    int e  = t >> 4;
    int c4 = t & 15;
    int s = ei[e];
    int d = ei[EE + e];
    float4 v = ((const float4*)x)[(size_t)s * 16 + c4];
    float* ap = agg + (size_t)d * HH + c4 * 4;
    unsafeAtomicAdd(ap + 0, v.x);
    unsafeAtomicAdd(ap + 1, v.y);
    unsafeAtomicAdd(ap + 2, v.z);
    unsafeAtomicAdd(ap + 3, v.w);
}

// ---------------------------------------------------------------------------
// fused GIN MLP: x = relu(relu((x+agg) @ W1 + b1) @ W2 + b2), one wave per 16-node tile
__global__ __launch_bounds__(256) void gin_mlp(
    const float* __restrict__ x, const float* __restrict__ agg,
    const __bf16* __restrict__ wfragL,     // layer base: gemm1 frags at +0, gemm2 at +4096
    const float* __restrict__ b1l, const float* __restrict__ b2l,
    float* __restrict__ xout, int ntiles)
{
    __shared__ __bf16 tiles[8][16 * HH];   // 16 KB; per-wave 16x64 bf16 staging
    const int wave = threadIdx.x >> 5;
    const int lane = threadIdx.x & 31;
    const int tile = blockIdx.x * 8 + wave;
    const bool active = tile < ntiles;
    __bf16* t = tiles[wave];
    const int half = lane >> 4;
    const int m = lane & 15;

    // stage 1: coalesced load of contiguous 16x64 tile of (x+agg), cast to bf16 in LDS
    if (active) {
        const float4* x4 = (const float4*)(x  + (size_t)tile * 1024);
        const float4* a4 = (const float4*)(agg + (size_t)tile * 1024);
#pragma unroll
        for (int j = 0; j < 8; ++j) {
            int idx = j * 32 + lane;
            float4 xv = x4[idx];
            float4 av = a4[idx];
            t[idx * 4 + 0] = (__bf16)(xv.x + av.x);
            t[idx * 4 + 1] = (__bf16)(xv.y + av.y);
            t[idx * 4 + 2] = (__bf16)(xv.z + av.z);
            t[idx * 4 + 3] = (__bf16)(xv.w + av.w);
        }
    }
    __syncthreads();

    v8f acc[4];
    // GEMM1: h = relu((x+agg) @ W1 + b1)
    if (active) {
        v16bf af0, af1;
#pragma unroll
        for (int i = 0; i < 8; ++i) {
            af0[i]     = t[m * HH +  0 +  8 * half + i];
            af0[8 + i] = t[m * HH + 16 +  8 * half + i];
            af1[i]     = t[m * HH + 32 +  8 * half + i];
            af1[8 + i] = t[m * HH + 48 +  8 * half + i];
        }
        const v16bf* bf = (const v16bf*)wfragL;
#pragma unroll
        for (int nt = 0; nt < 4; ++nt) {
            v8f c = {0.f, 0.f, 0.f, 0.f, 0.f, 0.f, 0.f, 0.f};
            v16bf b0 = bf[(nt * 2 + 0) * 32 + lane];
            v16bf b1 = bf[(nt * 2 + 1) * 32 + lane];
            c = __builtin_amdgcn_wmma_f32_16x16x32_bf16(false, af0, false, b0, (short)0, c, false, false);
            c = __builtin_amdgcn_wmma_f32_16x16x32_bf16(false, af1, false, b1, (short)0, c, false, false);
            float bb = b1l[nt * 16 + m];
#pragma unroll
            for (int r = 0; r < 8; ++r) {
                float v = c[r] + bb;
                c[r] = v > 0.f ? v : 0.f;
            }
            acc[nt] = c;
        }
    }
    __syncthreads();

    // stage 2: spill h (C/D layout: M = r + 8*half, N = 16*nt + m) to LDS as bf16 row-major tile
    if (active) {
#pragma unroll
        for (int nt = 0; nt < 4; ++nt)
#pragma unroll
            for (int r = 0; r < 8; ++r)
                t[(r + 8 * half) * HH + nt * 16 + m] = (__bf16)acc[nt][r];
    }
    __syncthreads();

    // GEMM2: x = relu(h @ W2 + b2), write fp32 back to global
    if (active) {
        v16bf af0, af1;
#pragma unroll
        for (int i = 0; i < 8; ++i) {
            af0[i]     = t[m * HH +  0 +  8 * half + i];
            af0[8 + i] = t[m * HH + 16 +  8 * half + i];
            af1[i]     = t[m * HH + 32 +  8 * half + i];
            af1[8 + i] = t[m * HH + 48 +  8 * half + i];
        }
        const v16bf* bf = (const v16bf*)(wfragL + 4096);
#pragma unroll
        for (int nt = 0; nt < 4; ++nt) {
            v8f c = {0.f, 0.f, 0.f, 0.f, 0.f, 0.f, 0.f, 0.f};
            v16bf b0 = bf[(nt * 2 + 0) * 32 + lane];
            v16bf b1 = bf[(nt * 2 + 1) * 32 + lane];
            c = __builtin_amdgcn_wmma_f32_16x16x32_bf16(false, af0, false, b0, (short)0, c, false, false);
            c = __builtin_amdgcn_wmma_f32_16x16x32_bf16(false, af1, false, b1, (short)0, c, false, false);
            float bb = b2l[nt * 16 + m];
#pragma unroll
            for (int r = 0; r < 8; ++r) {
                float v = c[r] + bb;
                xout[((size_t)tile * 16 + r + 8 * half) * HH + nt * 16 + m] = v > 0.f ? v : 0.f;
            }
        }
    }
}

// ---------------------------------------------------------------------------
// global add pool: g[batch[node]] += x[node]
__global__ void gin_pool(const float* __restrict__ x, const int* __restrict__ batch,
                         float* __restrict__ g) {
    int t = blockIdx.x * blockDim.x + threadIdx.x;  // N*H exact
    int node = t >> 6, c = t & 63;
    unsafeAtomicAdd(&g[(size_t)batch[node] * HH + c], x[t]);
}

// ---------------------------------------------------------------------------
// decoder: out = relu(g @ Wd1 + bd1) @ Wd2 + bd2 ; one block, one row per thread
__global__ __launch_bounds__(256) void gin_decoder(
    const float* __restrict__ g, const float* __restrict__ Wd1, const float* __restrict__ bd1,
    const float* __restrict__ Wd2, const float* __restrict__ bd2, float* __restrict__ out)
{
    __shared__ float w1s[HH * HH];
    __shared__ float w2s[HH * OUTD];
    __shared__ float b1s[HH];
    __shared__ float b2s[OUTD];
    int tid = threadIdx.x;
    for (int i = tid; i < HH * HH; i += 256) w1s[i] = Wd1[i];
    for (int i = tid; i < HH * OUTD; i += 256) w2s[i] = Wd2[i];
    if (tid < HH) b1s[tid] = bd1[tid];
    if (tid < OUTD) b2s[tid] = bd2[tid];
    __syncthreads();

    float h[HH];
#pragma unroll
    for (int n = 0; n < HH; ++n) h[n] = b1s[n];
    for (int k = 0; k < HH; ++k) {
        float gk = g[(size_t)tid * HH + k];
#pragma unroll
        for (int n = 0; n < HH; ++n) h[n] += gk * w1s[k * HH + n];
    }
#pragma unroll
    for (int n = 0; n < HH; ++n) h[n] = h[n] > 0.f ? h[n] : 0.f;
    for (int o = 0; o < OUTD; ++o) {
        float a = b2s[o];
#pragma unroll
        for (int n = 0; n < HH; ++n) a += h[n] * w2s[n * OUTD + o];
        out[tid * OUTD + o] = a;
    }
}

// ---------------------------------------------------------------------------
extern "C" void kernel_launch(void* const* d_in, const int* in_sizes, int n_in,
                              void* d_out, int out_size, void* d_ws, size_t ws_size,
                              hipStream_t stream) {
    const int*   edge_index = (const int*)d_in[0];
    const int*   batch      = (const int*)d_in[1];
    const float* W_enc      = (const float*)d_in[2];
    const float* b_enc      = (const float*)d_in[3];
    const float* W1         = (const float*)d_in[4];
    const float* b1         = (const float*)d_in[5];
    const float* W2         = (const float*)d_in[6];
    const float* b2         = (const float*)d_in[7];
    const float* Wd1        = (const float*)d_in[8];
    const float* bd1        = (const float*)d_in[9];
    const float* Wd2        = (const float*)d_in[10];
    const float* bd2        = (const float*)d_in[11];
    float* out = (float*)d_out;

    char* ws = (char*)d_ws;
    size_t off = 0;
    float* x    = (float*)(ws + off); off += (size_t)NN * HH * sizeof(float);
    float* agg  = (float*)(ws + off); off += (size_t)NN * HH * sizeof(float);
    float* deg  = (float*)(ws + off); off += ((size_t)NN * sizeof(float) + 255) & ~(size_t)255;
    float* g    = (float*)(ws + off); off += (size_t)GG * HH * sizeof(float);
    __bf16* wfrag = (__bf16*)(ws + off); off += (size_t)LL * 2 * HH * HH * sizeof(__bf16);
    (void)off; (void)ws_size; (void)in_sizes; (void)n_in; (void)out_size;

    const int ntiles = NN / 16;  // 6250

    // degree + encoder
    gin_fill0<<<(NN / 4 + 255) / 256, 256, 0, stream>>>((float4*)deg, NN / 4);
    gin_deg<<<(EE + 255) / 256, 256, 0, stream>>>(edge_index, deg);
    gin_enc<<<NN * HH / 256, 256, 0, stream>>>(deg, W_enc, b_enc, x);

    // weight fragment prep (all 12 layers, both GEMMs)
    gin_wprep<<<LL * 2 * HH * HH / 256, 256, 0, stream>>>(W1, W2, wfrag);

    // 12 GIN layers
    for (int l = 0; l < LL; ++l) {
        gin_fill0<<<NN * HH / 4 / 256, 256, 0, stream>>>((float4*)agg, NN * HH / 4);
        gin_scatter<<<EE * 16 / 256, 256, 0, stream>>>(edge_index, x, agg);
        gin_mlp<<<(ntiles + 7) / 8, 256, 0, stream>>>(
            x, agg, wfrag + (size_t)l * 2 * HH * HH, b1 + l * HH, b2 + l * HH, x, ntiles);
    }

    // pool + decoder
    gin_fill0<<<GG * HH / 4 / 256, 256, 0, stream>>>((float4*)g, GG * HH / 4);
    gin_pool<<<NN * HH / 256, 256, 0, stream>>>(x, batch, g);
    gin_decoder<<<1, 256, 0, stream>>>(g, Wd1, bd1, Wd2, bd2, out);
}
